// LstmCrf_24120536334519
// MI455X (gfx1250) — compile-verified, matrix operand order checked by reading, not run on hardware
//
#include <hip/hip_runtime.h>
#include <hip/hip_bf16.h>

// ---------------------------------------------------------------- constants
#define T_  512
#define B_  256
#define E_  300
#define EP  320      // E padded to multiple of 32 (zero-filled)
#define H_  256
#define G4  1024     // 4*H
#define HID 512
#define KT  9
#define KTP 16       // tags padded to one WMMA N-tile

#define LSTM_THREADS 512          // 16 waves (4 per SIMD); 4 N-tiles per wave
#define NT_PER_WAVE  4
#define KC_IN   (EP / 32)         // 10 input K-chunks
#define KC_HID  (H_ / 32)         // 8 hidden K-chunks
#define KC_ALL  (KC_IN + KC_HID)  // 18 chunks per N-tile
#define PIPE    4                 // software pipeline depth (load lookahead)

typedef __attribute__((ext_vector_type(8)))  float  v8f;
typedef __attribute__((ext_vector_type(8)))  __bf16 v8bf;
typedef __attribute__((ext_vector_type(16))) __bf16 v16bf;

// WMMA fragment load: lane holds 16 bf16 = K {kb..kb+7} and {kb+16..kb+23}
// (CDNA5 16-bit A/B layout; p must be 16B aligned).
__device__ __forceinline__ v16bf ld_frag(const __bf16* p) {
    union { v16bf v; v8bf h[2]; } u;
    u.h[0] = *(const v8bf*)(p);
    u.h[1] = *(const v8bf*)(p + 16);
    return u.v;
}

__device__ __forceinline__ v8f wmma_bf16(v16bf a, v16bf b, v8f c) {
    return __builtin_amdgcn_wmma_f32_16x16x32_bf16(
        /*neg_a=*/false, a, /*neg_b=*/false, b,
        /*c_mod=*/(short)0, c, /*reuse_a=*/false, /*reuse_b=*/false);
}

__device__ __forceinline__ float sigf(float x) { return 1.0f / (1.0f + __expf(-x)); }

// ---------------------------------------------------------------- kernel 0
// Convert/pad weights to bf16 once per call (tiny).
__global__ __launch_bounds__(256)
void prep_weights(const float* wihf, const float* wihb,
                  const float* whhf, const float* whhb,
                  const float* wout_f,
                  __bf16* wih, __bf16* whh, __bf16* wout)
{
    const int NWIH = 2 * G4 * EP;
    const int NWHH = 2 * G4 * H_;
    const int NWO  = KTP * HID;
    int total = NWIH + NWHH + NWO;
    for (int i = blockIdx.x * blockDim.x + threadIdx.x; i < total;
         i += gridDim.x * blockDim.x) {
        if (i < NWIH) {
            int d = i / (G4 * EP);
            int r = (i / EP) % G4;
            int k = i % EP;
            const float* src = d ? wihb : wihf;
            wih[i] = (__bf16)((k < E_) ? src[r * E_ + k] : 0.0f);
        } else if (i < NWIH + NWHH) {
            int j = i - NWIH;
            int d = j / (G4 * H_);
            int rem = j % (G4 * H_);
            const float* src = d ? whhb : whhf;
            whh[j] = (__bf16)src[rem];
        } else {
            int j = i - NWIH - NWHH;
            int r = j / HID, c = j % HID;
            wout[j] = (__bf16)((r < KT) ? wout_f[r * HID + c] : 0.0f);
        }
    }
}

// ---------------------------------------------------------------- kernel 1
// Bidirectional LSTM. grid = 32 workgroups: [dir(2)] x [batch tile(16)].
// Each WG owns 16 batch rows and walks all T steps privately (the recurrence
// is independent across batch rows).  16 waves (4/SIMD) each compute 4
// N-tiles of the 16x1024 gate matrix per step.  Weights stream from L2 every
// step (asm clobber stops LICM from hoisting/spilling them), and the K-loop
// is explicitly software-pipelined with a depth-4 rotating fragment buffer
// so each WMMA's wait leaves 3 load-pairs in flight (deep L2 overlap at a
// bounded ~64-VGPR cost).
__global__ __launch_bounds__(LSTM_THREADS)
void lstm_recurrent(const int* __restrict__ input_ids,
                    const int* __restrict__ seq_len,
                    const float* __restrict__ emb_table,
                    const __bf16* __restrict__ wih_all,
                    const __bf16* __restrict__ whh_all,
                    const float* __restrict__ b_f,
                    const float* __restrict__ b_b,
                    __bf16* __restrict__ hs_out)
{
    __shared__ __bf16 s_emb[16][EP];   // 10 KB  current embedding tile (bf16)
    __shared__ __bf16 s_hbf[16][H_];   //  8 KB  h carry as WMMA-A operand
    __shared__ float  s_hf [16][H_];   // 16 KB  h carry, f32
    __shared__ float  s_c  [16][H_];   // 16 KB  c carry, f32
    __shared__ float  s_g  [16][G4];   // 64 KB  gate accumulators
    __shared__ float  s_bias[G4];      //  4 KB
    __shared__ int    s_len[16];

    const int dir  = blockIdx.x >> 4;
    const int b0   = (blockIdx.x & 15) * 16;
    const int tid  = threadIdx.x;
    const int wave = tid >> 5;        // 0..15
    const int lane = tid & 31;
    const int lo   = lane & 15;       // row / col within 16-tile
    const int hi   = lane >> 4;       // selects K-half of the fragment

    const __bf16* Wih = wih_all + (size_t)dir * G4 * EP;
    const __bf16* Whh = whh_all + (size_t)dir * G4 * H_;
    const float*  bias = dir ? b_b : b_f;

    if (tid < 16) s_len[tid] = seq_len[b0 + tid];
    for (int i = tid; i < G4; i += LSTM_THREADS) s_bias[i] = bias[i];
    for (int i = tid; i < 16 * H_; i += LSTM_THREADS) {
        (&s_hf[0][0])[i] = 0.0f;
        (&s_c [0][0])[i] = 0.0f;
        (&s_hbf[0][0])[i] = (__bf16)0.0f;
    }
    __syncthreads();

    for (int step = 0; step < T_; ++step) {
        const int t = dir ? (T_ - 1 - step) : step;

        // Stop LICM from hoisting the (loop-invariant) weight fragment loads
        // out of the time loop -- hoisting them spills to scratch and puts
        // scratch latency on the recurrence critical path.
        asm volatile("" ::: "memory");

        // ---- stage 1: gather + cvt embeddings for this step's 16 rows ----
        for (int i = tid; i < 16 * EP; i += LSTM_THREADS) {
            int r = i / EP, k = i - r * EP;
            int token = input_ids[t * B_ + b0 + r];
            float v = (k < E_) ? emb_table[(size_t)token * E_ + k] : 0.0f;
            s_emb[r][k] = (__bf16)v;
        }
        __syncthreads();

        // ---- stage 2: gates(16x1024) = emb@Wih^T + h@Whh^T (WMMA bf16) ----
        #pragma unroll
        for (int nt = 0; nt < NT_PER_WAVE; ++nt) {
            const int n = wave * NT_PER_WAVE + nt;
            const __bf16* wrow_i = Wih + (size_t)(n * 16 + lo) * EP;
            const __bf16* wrow_h = Whh + (size_t)(n * 16 + lo) * H_;

            // unified K-chunk fragment loaders (kc: 0..9 input, 10..17 hidden)
            auto ldA = [&](int kc) -> v16bf {
                return (kc < KC_IN)
                    ? ld_frag(&s_emb[lo][kc * 32 + hi * 8])
                    : ld_frag(&s_hbf[lo][(kc - KC_IN) * 32 + hi * 8]);
            };
            auto ldB = [&](int kc) -> v16bf {
                return (kc < KC_IN)
                    ? ld_frag(wrow_i + kc * 32 + hi * 8)
                    : ld_frag(wrow_h + (kc - KC_IN) * 32 + hi * 8);
            };

            // depth-PIPE software pipeline: loads run PIPE chunks ahead of
            // the WMMA that consumes them.
            v16bf aq[PIPE], bq[PIPE];
            #pragma unroll
            for (int p = 0; p < PIPE; ++p) { aq[p] = ldA(p); bq[p] = ldB(p); }

            v8f acc = {};
            #pragma unroll
            for (int kc = 0; kc < KC_ALL; ++kc) {
                const int s = kc & (PIPE - 1);
                acc = wmma_bf16(aq[s], bq[s], acc);
                if (kc + PIPE < KC_ALL) {
                    aq[s] = ldA(kc + PIPE);
                    bq[s] = ldB(kc + PIPE);
                }
            }

            #pragma unroll
            for (int r = 0; r < 8; ++r)         // C layout: M = r + 8*hi, N = lo
                s_g[r + 8 * hi][n * 16 + lo] = acc[r];
        }
        __syncthreads();

        // ---- stage 3: elementwise LSTM cell + length masking ----
        {
            const int col   = tid & 255;        // hidden column
            const int rbase = (tid >> 8) * 8;   // rows 0-7 or 8-15
            #pragma unroll
            for (int q = 0; q < 8; ++q) {
                const int r = rbase + q;
                float gi = s_g[r][          col] + s_bias[          col];
                float gf = s_g[r][H_      + col] + s_bias[H_      + col];
                float gg = s_g[r][2 * H_  + col] + s_bias[2 * H_  + col];
                float go = s_g[r][3 * H_  + col] + s_bias[3 * H_  + col];
                float cold = s_c[r][col];
                float hold = s_hf[r][col];
                float cn = sigf(gf) * cold + sigf(gi) * tanhf(gg);
                float hn = sigf(go) * tanhf(cn);
                bool  m  = (t < s_len[r]);
                float hsel = m ? hn : hold;
                float csel = m ? cn : cold;
                s_c [r][col] = csel;
                s_hf[r][col] = hsel;
                __bf16 hb = (__bf16)hsel;
                s_hbf[r][col] = hb;
                hs_out[((size_t)t * B_ + b0 + r) * HID + dir * H_ + col] = hb;
            }
        }
        __syncthreads();
    }
}

// ---------------------------------------------------------------- kernel 2
// emissions = lstm_out @ W_out^T + b_out.  M = T*B, K = 512, N = 16 (9 used).
// One wave per 16-row M-tile; stored f32 padded to 16 tags.
__global__ __launch_bounds__(256)
void emissions_kernel(const __bf16* __restrict__ hs,
                      const __bf16* __restrict__ wout,
                      const float*  __restrict__ b_out,
                      float* __restrict__ emis)
{
    const int wave = threadIdx.x >> 5;
    const int lane = threadIdx.x & 31;
    const int lo = lane & 15, hi = lane >> 4;
    const int mt = blockIdx.x * 8 + wave;              // 0 .. T*B/16 - 1
    const size_t arow = (size_t)mt * 16 + lo;

    v8f acc = {};
    const __bf16* ap = hs + arow * HID;
    const __bf16* bp = wout + (size_t)lo * HID;
    #pragma unroll
    for (int kc = 0; kc < HID / 32; ++kc) {
        int kb = kc * 32 + hi * 8;
        v16bf a = ld_frag(ap + kb);
        v16bf b = ld_frag(bp + kb);
        acc = wmma_bf16(a, b, acc);
    }
    float bo = (lo < KT) ? b_out[lo] : 0.0f;
    #pragma unroll
    for (int r = 0; r < 8; ++r) {
        size_t orow = (size_t)mt * 16 + r + 8 * hi;
        emis[orow * KTP + lo] = acc[r] + bo;
    }
}

// ---------------------------------------------------------------- kernel 3
// CRF: numerator, log-partition (forward), Viterbi + backtrace, loss.
// One lane per batch element; K=9 state vectors live in registers.
__global__ __launch_bounds__(256)
void crf_kernel(const float* __restrict__ emis,
                const int*   __restrict__ labels,
                const int*   __restrict__ seq_len,
                const float* __restrict__ start_tr,
                const float* __restrict__ end_tr,
                const float* __restrict__ trans,
                unsigned char* __restrict__ hist,
                float* __restrict__ d_out)
{
    __shared__ float s_tr[KT * KT];
    __shared__ float s_st[KT], s_en[KT];
    __shared__ float s_loss[B_];

    const int b = threadIdx.x;
    if (b < KT * KT) s_tr[b] = trans[b];
    if (b < KT) { s_st[b] = start_tr[b]; s_en[b] = end_tr[b]; }
    __syncthreads();

    const int len = seq_len[b];
    float alpha[KT], score[KT], em[KT];

    #pragma unroll
    for (int j = 0; j < KT; ++j) em[j] = emis[(size_t)b * KTP + j];
    int y0 = labels[b];
    float num = s_st[y0] + em[y0];
    #pragma unroll
    for (int j = 0; j < KT; ++j) { alpha[j] = s_st[j] + em[j]; score[j] = alpha[j]; }

    for (int t = 1; t < T_; ++t) {
        #pragma unroll
        for (int j = 0; j < KT; ++j) em[j] = emis[((size_t)t * B_ + b) * KTP + j];
        const bool m = (t < len);
        int yp = labels[(t - 1) * B_ + b];
        int yc = labels[t * B_ + b];
        if (m) num += s_tr[yp * KT + yc] + em[yc];

        float na[KT], ns[KT];
        #pragma unroll
        for (int j = 0; j < KT; ++j) {
            float mx = -1e30f, bmx = -1e30f;
            int   bi = 0;
            #pragma unroll
            for (int i = 0; i < KT; ++i) {
                float tv = s_tr[i * KT + j];
                float av = alpha[i] + tv;
                mx = fmaxf(mx, av);
                float sv = score[i] + tv;
                if (sv > bmx) { bmx = sv; bi = i; }
            }
            float s = 0.0f;
            #pragma unroll
            for (int i = 0; i < KT; ++i)
                s += __expf(alpha[i] + s_tr[i * KT + j] - mx);
            na[j] = mx + __logf(s) + em[j];
            ns[j] = bmx + em[j];
            hist[((size_t)(t - 1) * B_ + b) * KT + j] = (unsigned char)bi;
        }
        if (m) {
            #pragma unroll
            for (int j = 0; j < KT; ++j) { alpha[j] = na[j]; score[j] = ns[j]; }
        }
    }

    // denominator + numerator end term
    float mx = -1e30f;
    #pragma unroll
    for (int j = 0; j < KT; ++j) mx = fmaxf(mx, alpha[j] + s_en[j]);
    float s = 0.0f;
    #pragma unroll
    for (int j = 0; j < KT; ++j) s += __expf(alpha[j] + s_en[j] - mx);
    float denom = mx + __logf(s);
    num += s_en[labels[(len - 1) * B_ + b]];

    s_loss[b] = num - denom;
    __syncthreads();
    for (int off = B_ / 2; off > 0; off >>= 1) {
        if (b < off) s_loss[b] += s_loss[b + off];
        __syncthreads();
    }
    if (b == 0) d_out[0] = -s_loss[0];

    // Viterbi backtrace
    int   bl = 0;
    float bs = score[0] + s_en[0];
    #pragma unroll
    for (int j = 1; j < KT; ++j) {
        float v = score[j] + s_en[j];
        if (v > bs) { bs = v; bl = j; }
    }
    int tag = bl;
    d_out[1 + (size_t)(T_ - 1) * B_ + b] = ((T_ - 1) < len) ? (float)bl : 0.0f;
    for (int t = T_ - 2; t >= 0; --t) {
        bool mn = ((t + 1) < len);
        tag = mn ? (int)hist[((size_t)t * B_ + b) * KT + tag] : bl;
        d_out[1 + (size_t)t * B_ + b] = (t < len) ? (float)tag : 0.0f;
    }
}

// ---------------------------------------------------------------- launcher
extern "C" void kernel_launch(void* const* d_in, const int* in_sizes, int n_in,
                              void* d_out, int out_size, void* d_ws, size_t ws_size,
                              hipStream_t stream)
{
    const int*   input_ids  = (const int*)  d_in[0];
    const int*   seq_len    = (const int*)  d_in[1];
    const int*   label_ids  = (const int*)  d_in[2];
    const float* emb_table  = (const float*)d_in[3];
    const float* W_ih_f     = (const float*)d_in[4];
    const float* W_hh_f     = (const float*)d_in[5];
    const float* b_f        = (const float*)d_in[6];
    const float* W_ih_b     = (const float*)d_in[7];
    const float* W_hh_b     = (const float*)d_in[8];
    const float* b_b        = (const float*)d_in[9];
    const float* W_out      = (const float*)d_in[10];
    const float* b_out      = (const float*)d_in[11];
    const float* start_tr   = (const float*)d_in[12];
    const float* end_tr     = (const float*)d_in[13];
    const float* trans      = (const float*)d_in[14];
    float* out = (float*)d_out;

    // workspace carve-up (256B aligned)
    char* ws = (char*)d_ws;
    size_t o = 0;
    auto take = [&](size_t bytes) { char* p = ws + o; o = (o + bytes + 255) & ~(size_t)255; return p; };
    __bf16* wih  = (__bf16*)take((size_t)2 * G4 * EP * 2);          // 1.25 MB
    __bf16* whh  = (__bf16*)take((size_t)2 * G4 * H_ * 2);          // 1.0 MB
    __bf16* wout = (__bf16*)take((size_t)KTP * HID * 2);            // 16 KB
    __bf16* hs   = (__bf16*)take((size_t)T_ * B_ * HID * 2);        // 134 MB
    float*  emis = (float*) take((size_t)T_ * B_ * KTP * 4);        // 8.4 MB
    unsigned char* hist = (unsigned char*)take((size_t)(T_ - 1) * B_ * KT); // 1.2 MB
    (void)ws_size; (void)in_sizes; (void)n_in; (void)out_size;

    {
        int total = 2 * G4 * EP + 2 * G4 * H_ + KTP * HID;
        int blocks = (total + 255) / 256;
        prep_weights<<<blocks, 256, 0, stream>>>(W_ih_f, W_ih_b, W_hh_f, W_hh_b,
                                                 W_out, wih, whh, wout);
    }
    lstm_recurrent<<<32, LSTM_THREADS, 0, stream>>>(input_ids, seq_len, emb_table,
                                                    wih, whh, b_f, b_b, hs);
    emissions_kernel<<<(T_ * B_ / 16) / 8, 256, 0, stream>>>(hs, wout, b_out, emis);
    crf_kernel<<<1, B_, 0, stream>>>(emis, label_ids, seq_len, start_tr, end_tr,
                                     trans, hist, out);
}
